// GAT_87050397155997
// MI455X (gfx1250) — compile-verified, matrix operand order checked by reading, not run on hardware
//
#include <hip/hip_runtime.h>
#include <hip/hip_bf16.h>

// ---------------------------------------------------------------------------
// Types for CDNA5 WMMA
// ---------------------------------------------------------------------------
typedef __attribute__((ext_vector_type(16))) __bf16 v16bf;
typedef __attribute__((ext_vector_type(8)))  __bf16 v8bf;
typedef __attribute__((ext_vector_type(8)))  float  v8f;

struct Q2 { uint4 a, b; };   // 32 bytes == one v16bf fragment

// pack 8 consecutive floats into 8 bf16 (one uint4) via native converts
// (clang lowers these casts to gfx1250 packed bf16 cvt instructions)
__device__ __forceinline__ uint4 pack8(const float* f) {
    v8bf h;
    #pragma unroll
    for (int i = 0; i < 8; ++i) h[i] = (__bf16)f[i];
    return __builtin_bit_cast(uint4, h);
}
// K-group permutation: groups of 8 at base {0,8,16,24} -> {0,16,8,24}
__device__ __forceinline__ int slotbase(int b) {
    return (b == 8) ? 16 : (b == 16) ? 8 : b;
}

// monotonic float <-> uint encoding for atomicMax on signed floats
__device__ __forceinline__ unsigned encf(float f) {
    unsigned b = __float_as_uint(f);
    return (b & 0x80000000u) ? ~b : (b | 0x80000000u);
}
__device__ __forceinline__ float decf(unsigned k) {
    unsigned b = (k & 0x80000000u) ? (k & 0x7FFFFFFFu) : ~k;
    return __uint_as_float(b);
}

// ---------------------------------------------------------------------------
// GEMM: Y[M,Nc] = X[M,K] @ W[K,Nc] (+ bias), f32 in/out, bf16 WMMA compute.
// Block = 256 threads (8 waves). Tile: BM=128, BN=64, BK=32.
// Wave w computes rows w*16..w*16+15 x all 64 cols (4 WMMA accumulators).
// LDS tiles stored K-permuted so each lane's fragment is 32 contiguous bytes.
// Ping-pong double buffered; register-staged global loads.
// Requires: K % 32 == 0, Nc % 64 == 0 (true for all launches here).
// ---------------------------------------------------------------------------
__global__ __launch_bounds__(256)
void gemm_bf16wmma_kernel(const float* __restrict__ X,
                          const float* __restrict__ Wm,
                          const float* __restrict__ bias,
                          float* __restrict__ Y,
                          int M, int K, int Nc) {
    const int LDA = 40;                       // halves per A row (80B, aligned, no bank clash)
    const int LDB = 40;                       // halves per B column
    __shared__ unsigned short sA[2][128 * LDA];
    __shared__ unsigned short sB[2][64 * LDB];

    const int tid    = threadIdx.x;
    const int wave   = tid >> 5;
    const int lane   = tid & 31;
    const int lane16 = lane & 15;
    const int hi     = lane >> 4;

    const int rowBase = blockIdx.y * 128;
    const int colBase = blockIdx.x * 64;

    // staging assignments
    const int arow = tid >> 1;                // 0..127
    const int acg  = (tid & 1) * 16;          // col base 0 or 16 (16 floats / thread)
    const int bcol = tid & 63;                // 0..63
    const int bk0  = (tid >> 6) * 8;          // 0,8,16,24 (8 k / thread)

    const bool aRowOk = (rowBase + arow) < M;

    float4 ax[4];                             // 16 floats of A row
    float  bx[8];                             // 8 strided floats of one W column

    auto loadA = [&](int kt) {
        if (aRowOk) {
            const float4* src = (const float4*)(X + (size_t)(rowBase + arow) * K + kt + acg);
            ax[0] = src[0]; ax[1] = src[1]; ax[2] = src[2]; ax[3] = src[3];
        } else {
            ax[0] = ax[1] = ax[2] = ax[3] = make_float4(0.f, 0.f, 0.f, 0.f);
        }
    };
    auto loadB = [&](int kt) {
        const float* src = Wm + (size_t)(kt + bk0) * Nc + colBase + bcol;
        #pragma unroll
        for (int j = 0; j < 8; ++j) bx[j] = src[(size_t)j * Nc];
    };
    auto storeAB = [&](int buf) {
        *(uint4*)&sA[buf][arow * LDA + slotbase(acg)]     = pack8((const float*)&ax[0]);
        *(uint4*)&sA[buf][arow * LDA + slotbase(acg + 8)] = pack8((const float*)&ax[2]);
        *(uint4*)&sB[buf][bcol * LDB + slotbase(bk0)]     = pack8(bx);
    };

    v8f acc[4] = {v8f{}, v8f{}, v8f{}, v8f{}};

    loadA(0); loadB(0);
    storeAB(0);
    __syncthreads();

    int p = 0;
    for (int kt = 0; kt < K; kt += 32) {
        const bool hasNext = (kt + 32) < K;
        if (hasNext) { loadA(kt + 32); loadB(kt + 32); }

        // A fragment: rows wave*16 + lane16, contiguous 32B at half-offset hi*16
        Q2 qa;
        const unsigned short* ab = &sA[p][(wave * 16 + lane16) * LDA + hi * 16];
        qa.a = *(const uint4*)(ab);
        qa.b = *(const uint4*)(ab + 8);
        v16bf afrag = __builtin_bit_cast(v16bf, qa);

        #pragma unroll
        for (int g = 0; g < 4; ++g) {
            Q2 qb;
            const unsigned short* bb = &sB[p][(g * 16 + lane16) * LDB + hi * 16];
            qb.a = *(const uint4*)(bb);
            qb.b = *(const uint4*)(bb + 8);
            v16bf bfrag = __builtin_bit_cast(v16bf, qb);
            acc[g] = __builtin_amdgcn_wmma_f32_16x16x32_bf16(
                false, afrag, false, bfrag, (short)0, acc[g], false, false);
        }

        if (hasNext) {
            storeAB(p ^ 1);
            __syncthreads();
            p ^= 1;
        }
    }

    // epilogue: C/D layout -> M = r + 8*hi, N = lane16
    #pragma unroll
    for (int g = 0; g < 4; ++g) {
        const int gc = colBase + g * 16 + lane16;
        const float bv = bias ? bias[gc] : 0.0f;
        #pragma unroll
        for (int r = 0; r < 8; ++r) {
            int gm = rowBase + wave * 16 + r + 8 * hi;
            if (gm < M) Y[(size_t)gm * Nc + gc] = acc[g][r] + bv;
        }
    }
}

// ---------------------------------------------------------------------------
// Utility: zero a u32 buffer
// ---------------------------------------------------------------------------
__global__ void zero_u32_kernel(unsigned* __restrict__ p, long long n) {
    long long i = (long long)blockIdx.x * blockDim.x + threadIdx.x;
    if (i < n) p[i] = 0u;
}

// ---------------------------------------------------------------------------
// Edge pass 1: per-edge per-head GATv2 score; atomicMax into per-dst max.
// One wave per edge (incl. self-loops for e >= E).
// ---------------------------------------------------------------------------
__global__ __launch_bounds__(256)
void edge_score_kernel(const int* __restrict__ ei, int E, int EA,
                       const float* __restrict__ xl, const float* __restrict__ xr,
                       const float* __restrict__ att,
                       float* __restrict__ p, unsigned* __restrict__ nmax,
                       int Hh, int C) {
    int wid  = (blockIdx.x * blockDim.x + threadIdx.x) >> 5;
    int lane = threadIdx.x & 31;
    if (wid >= EA) return;
    int src, dst;
    if (wid < E) { src = ei[wid]; dst = ei[E + wid]; }
    else         { src = dst = wid - E; }
    const int HC = Hh * C;
    const float* xs = xl + (size_t)src * HC;
    const float* xd = xr + (size_t)dst * HC;
    for (int h = 0; h < Hh; ++h) {
        float s = 0.0f;
        for (int c = lane; c < C; c += 32) {
            float z = xs[h * C + c] + xd[h * C + c];
            z = (z > 0.0f) ? z : 0.2f * z;                 // leaky_relu(0.2)
            s += att[h * C + c] * z;
        }
        #pragma unroll
        for (int o = 16; o > 0; o >>= 1) s += __shfl_xor(s, o);
        if (lane == 0) {
            p[(size_t)wid * Hh + h] = s;
            atomicMax(&nmax[(size_t)dst * Hh + h], encf(s));
        }
    }
}

// ---------------------------------------------------------------------------
// Edge pass 2: p = exp(s - max[dst]); accumulate per-dst sum.
// ---------------------------------------------------------------------------
__global__ void edge_exp_kernel(const int* __restrict__ ei, int E, int EA,
                                float* __restrict__ p,
                                const unsigned* __restrict__ nmax,
                                float* __restrict__ nsum, int Hh) {
    long long t = (long long)blockIdx.x * blockDim.x + threadIdx.x;
    if (t >= (long long)EA * Hh) return;
    int e = (int)(t / Hh), h = (int)(t % Hh);
    int dst = (e < E) ? ei[E + e] : (e - E);
    float pv = expf(p[t] - decf(nmax[(size_t)dst * Hh + h]));
    p[t] = pv;
    atomicAdd(&nsum[(size_t)dst * Hh + h], pv);
}

// ---------------------------------------------------------------------------
// Edge pass 3: agg[dst] += p * xl[src]  (numerator of softmax-weighted sum)
// ---------------------------------------------------------------------------
__global__ __launch_bounds__(256)
void edge_agg_kernel(const int* __restrict__ ei, int E, int EA,
                     const float* __restrict__ xl, const float* __restrict__ p,
                     float* __restrict__ agg, int Hh, int C) {
    int wid  = (blockIdx.x * blockDim.x + threadIdx.x) >> 5;
    int lane = threadIdx.x & 31;
    if (wid >= EA) return;
    int src, dst;
    if (wid < E) { src = ei[wid]; dst = ei[E + wid]; }
    else         { src = dst = wid - E; }
    const int HC = Hh * C;
    const float* xs = xl + (size_t)src * HC;
    float* ad = agg + (size_t)dst * HC;
    for (int h = 0; h < Hh; ++h) {
        float pv = p[(size_t)wid * Hh + h];
        for (int c = lane; c < C; c += 32)
            atomicAdd(&ad[h * C + c], pv * xs[h * C + c]);
    }
}

// ---------------------------------------------------------------------------
// Node epilogue (layer 1): h = elu(agg / sum + bias)
// ---------------------------------------------------------------------------
__global__ void node_finish_kernel(const float* __restrict__ agg,
                                   const float* __restrict__ nsum,
                                   const float* __restrict__ bias,
                                   float* __restrict__ out,
                                   int N, int Hh, int C) {
    long long t = (long long)blockIdx.x * blockDim.x + threadIdx.x;
    const int HC = Hh * C;
    if (t >= (long long)N * HC) return;
    int n = (int)(t / HC);
    int rem = (int)(t % HC);
    int h = rem / C;
    float v = agg[t] / (nsum[(size_t)n * Hh + h] + 1e-16f) + bias[rem];
    out[t] = (v > 0.0f) ? v : (expf(v) - 1.0f);            // elu
}

// ---------------------------------------------------------------------------
// Final: v = agg/sum + b2, then row-wise log_softmax over 64 classes.
// One wave per node, 2 classes per lane, shfl reductions (wave32).
// ---------------------------------------------------------------------------
__global__ __launch_bounds__(256)
void final_logsoftmax_kernel(const float* __restrict__ agg,
                             const float* __restrict__ nsum,
                             const float* __restrict__ bias,
                             float* __restrict__ out, int N) {
    int wid  = (blockIdx.x * blockDim.x + threadIdx.x) >> 5;
    int lane = threadIdx.x & 31;
    if (wid >= N) return;
    float den = nsum[wid] + 1e-16f;
    float v0 = agg[(size_t)wid * 64 + lane]      / den + bias[lane];
    float v1 = agg[(size_t)wid * 64 + 32 + lane] / den + bias[32 + lane];
    float m = fmaxf(v0, v1);
    #pragma unroll
    for (int o = 16; o > 0; o >>= 1) m = fmaxf(m, __shfl_xor(m, o));
    float s = expf(v0 - m) + expf(v1 - m);
    #pragma unroll
    for (int o = 16; o > 0; o >>= 1) s += __shfl_xor(s, o);
    float l = logf(s);
    out[(size_t)wid * 64 + lane]      = v0 - m - l;
    out[(size_t)wid * 64 + 32 + lane] = v1 - m - l;
}

// ---------------------------------------------------------------------------
// Host launcher
// ---------------------------------------------------------------------------
extern "C" void kernel_launch(void* const* d_in, const int* in_sizes, int n_in,
                              void* d_out, int out_size, void* d_ws, size_t ws_size,
                              hipStream_t stream) {
    const float* x    = (const float*)d_in[0];
    const int*   ei   = (const int*)  d_in[1];
    const float* Wl1  = (const float*)d_in[2];
    const float* Wr1  = (const float*)d_in[3];
    const float* bl1  = (const float*)d_in[4];
    const float* att1 = (const float*)d_in[5];
    const float* b1   = (const float*)d_in[6];
    const float* Wl2  = (const float*)d_in[7];
    const float* Wr2  = (const float*)d_in[8];
    const float* bl2  = (const float*)d_in[9];
    const float* att2 = (const float*)d_in[10];
    const float* b2   = (const float*)d_in[11];

    const int DIN = 512, HDH = 512, DOUT = 64, H1 = 8, C1 = 64;
    const int N  = in_sizes[0] / DIN;
    const int E  = in_sizes[1] / 2;
    const int EA = E + N;                     // edges + self-loops

    char* w = (char*)d_ws;
    size_t off = 0;
    auto alloc = [&](size_t bytes) -> void* {
        void* p = w + off;
        off = (off + bytes + 255) & ~(size_t)255;
        return p;
    };
    float*    xl1   = (float*)   alloc((size_t)N * HDH * 4);
    float*    xr1   = (float*)   alloc((size_t)N * HDH * 4);
    float*    hbuf  = (float*)   alloc((size_t)N * HDH * 4);   // agg1 -> h1 (in place)
    float*    p1    = (float*)   alloc((size_t)EA * H1 * 4);   // reused as p2
    unsigned* nmax  = (unsigned*)alloc((size_t)N * H1 * 4);    // reused layer 2
    float*    nsum  = (float*)   alloc((size_t)N * H1 * 4);    // reused layer 2
    float*    xl2   = (float*)   alloc((size_t)N * DOUT * 4);
    float*    xr2   = (float*)   alloc((size_t)N * DOUT * 4);
    float*    agg2  = (float*)   alloc((size_t)N * DOUT * 4);

    const int rowBlocks = (N + 127) / 128;
    const dim3 blk(256);

    // ---- Layer 1 GEMMs (bf16 WMMA, f32 accumulate) ----
    gemm_bf16wmma_kernel<<<dim3(HDH / 64, rowBlocks), blk, 0, stream>>>(
        x, Wl1, bl1, xl1, N, DIN, HDH);
    gemm_bf16wmma_kernel<<<dim3(HDH / 64, rowBlocks), blk, 0, stream>>>(
        x, Wr1, nullptr, xr1, N, DIN, HDH);

    // ---- Layer 1 segment softmax + aggregation ----
    {
        long long nz;
        nz = (long long)N * H1;
        zero_u32_kernel<<<(unsigned)((nz + 255) / 256), blk, 0, stream>>>(nmax, nz);
        zero_u32_kernel<<<(unsigned)((nz + 255) / 256), blk, 0, stream>>>((unsigned*)nsum, nz);
        nz = (long long)N * HDH;
        zero_u32_kernel<<<(unsigned)((nz + 255) / 256), blk, 0, stream>>>((unsigned*)hbuf, nz);

        int waveBlocks = (EA * 32 + 255) / 256;
        edge_score_kernel<<<waveBlocks, blk, 0, stream>>>(ei, E, EA, xl1, xr1, att1,
                                                          p1, nmax, H1, C1);
        long long te = (long long)EA * H1;
        edge_exp_kernel<<<(unsigned)((te + 255) / 256), blk, 0, stream>>>(ei, E, EA, p1,
                                                                          nmax, nsum, H1);
        edge_agg_kernel<<<waveBlocks, blk, 0, stream>>>(ei, E, EA, xl1, p1, hbuf, H1, C1);

        long long tn = (long long)N * HDH;
        node_finish_kernel<<<(unsigned)((tn + 255) / 256), blk, 0, stream>>>(
            hbuf, nsum, b1, hbuf, N, H1, C1);
    }

    // ---- Layer 2 GEMMs ----
    gemm_bf16wmma_kernel<<<dim3(DOUT / 64, rowBlocks), blk, 0, stream>>>(
        hbuf, Wl2, bl2, xl2, N, HDH, DOUT);
    gemm_bf16wmma_kernel<<<dim3(DOUT / 64, rowBlocks), blk, 0, stream>>>(
        hbuf, Wr2, nullptr, xr2, N, HDH, DOUT);

    // ---- Layer 2 segment softmax + aggregation + log_softmax ----
    {
        long long nz;
        nz = (long long)N;
        zero_u32_kernel<<<(unsigned)((nz + 255) / 256), blk, 0, stream>>>(nmax, nz);
        zero_u32_kernel<<<(unsigned)((nz + 255) / 256), blk, 0, stream>>>((unsigned*)nsum, nz);
        nz = (long long)N * DOUT;
        zero_u32_kernel<<<(unsigned)((nz + 255) / 256), blk, 0, stream>>>((unsigned*)agg2, nz);

        int waveBlocks = (EA * 32 + 255) / 256;
        edge_score_kernel<<<waveBlocks, blk, 0, stream>>>(ei, E, EA, xl2, xr2, att2,
                                                          p1, nmax, 1, DOUT);
        long long te = (long long)EA;
        edge_exp_kernel<<<(unsigned)((te + 255) / 256), blk, 0, stream>>>(ei, E, EA, p1,
                                                                          nmax, nsum, 1);
        edge_agg_kernel<<<waveBlocks, blk, 0, stream>>>(ei, E, EA, xl2, p1, agg2, 1, DOUT);

        int nodeWaveBlocks = (N * 32 + 255) / 256;
        final_logsoftmax_kernel<<<nodeWaveBlocks, blk, 0, stream>>>(agg2, nsum, b2,
                                                                   (float*)d_out, N);
    }
}
